// CL_Model_37520834297886
// MI455X (gfx1250) — matrix operand order repaired and checked by reading, not compiled, not run on hardware
//
#include <hip/hip_runtime.h>
#include <hip/hip_bf16.h>
#include <math.h>

// ---------------------------------------------------------------------------
// MI455X (gfx1250) implementation.
//
// Roofline: output = 2056*65537 fp32 = 539 MB stores -> ~23 us floor @23.3TB/s.
// Big GEMM (2048x128 @ 128x65536) = 34.4 GFLOP -> few us with f16 WMMA
// (v_wmma_f32_16x16x32_f16), so the whole thing is store-bandwidth bound.
// queue_var (32 MB) lives in the 192 MB L2 across tile re-reads; no LDS tiling
// needed. A/B operands are pre-packed into WMMA fragment lane order so every
// fragment load is a coalesced 32B-per-lane contiguous read.
// ---------------------------------------------------------------------------

#define DIMC   128
#define KQ     65536
#define SS     16
#define OUTW   65537          // 1 pos column + 65536 negatives
#define INV_T  5.0f           // 1 / T, T = 0.2

typedef __attribute__((ext_vector_type(16))) _Float16 v16h;
typedef __attribute__((ext_vector_type(8)))  float    v8f;

// ---------------- inv branch -------------------------------------------------

__global__ __launch_bounds__(128) void inv_prep(const float* __restrict__ qi,
                                                const float* __restrict__ ki,
                                                float* __restrict__ qn_ws,
                                                float* __restrict__ out) {
    const int b = blockIdx.x;
    const int c = threadIdx.x;
    const float q = qi[b * DIMC + c];
    const float k = ki[b * DIMC + c];
    __shared__ float s1[128], s2[128];
    s1[c] = q * q; s2[c] = k * k;
    __syncthreads();
    for (int off = 64; off > 0; off >>= 1) {
        if (c < off) { s1[c] += s1[c + off]; s2[c] += s2[c + off]; }
        __syncthreads();
    }
    const float qn = q / fmaxf(sqrtf(s1[0]), 1e-12f);
    const float kn = k / fmaxf(sqrtf(s2[0]), 1e-12f);
    __syncthreads();
    s1[c] = qn * kn;
    __syncthreads();
    for (int off = 64; off > 0; off >>= 1) {
        if (c < off) s1[c] += s1[c + off];
        __syncthreads();
    }
    if (c == 0) out[(size_t)b * OUTW] = s1[0] * INV_T;
    qn_ws[b * DIMC + c] = qn;
}

__global__ __launch_bounds__(256) void inv_neg(const float* __restrict__ qn,
                                               const float* __restrict__ queue,
                                               float* __restrict__ out) {
    __shared__ float sq[8 * DIMC];
    for (int t = threadIdx.x; t < 8 * DIMC; t += 256) sq[t] = qn[t];
    __syncthreads();
    const int j = blockIdx.x * 256 + threadIdx.x;
    float acc[8] = {0.f, 0.f, 0.f, 0.f, 0.f, 0.f, 0.f, 0.f};
    for (int d = 0; d < DIMC; ++d) {
        const float col = queue[(size_t)d * KQ + j];
#pragma unroll
        for (int b = 0; b < 8; ++b) acc[b] += sq[b * DIMC + d] * col;
    }
#pragma unroll
    for (int b = 0; b < 8; ++b)
        out[(size_t)b * OUTW + 1 + j] = acc[b] * INV_T;
}

// ---------------- pack queue_var into WMMA B-fragment layout -----------------
// B fragment (32K x 16N per wmma): lanes 0-15 hold K=0..15 of column n=lane,
// lanes 16-31 hold K=16..31 of column n=lane-16; half index i = k within group.
__global__ __launch_bounds__(256) void pack_queue(const float* __restrict__ queue,
                                                  _Float16* __restrict__ Bpack) {
    const int tid = blockIdx.x * 256 + threadIdx.x;   // k * 65536 + n
    const int n = tid & (KQ - 1);
    const int k = tid >> 16;
    const float v = queue[tid];
    const int kb = k >> 5, kk = k & 31;
    const int g = kk >> 4, i = kk & 15;
    const int nb = n >> 4;
    const int lane = (n & 15) + (g << 4);
    Bpack[((((nb << 2) + kb) << 5) + lane) * 16 + i] = (_Float16)v;
}

// ---------------- var branch: flip + crop + bilinear + l2norm ----------------

__device__ __forceinline__ float sample_map(const float* __restrict__ base, int flip,
                                            float lo_h, float hi_h,
                                            float lo_w, float hi_w,
                                            int sy, int sx, int* validOut) {
    const int h0 = (int)floorf(32.f * lo_h + 0.5f);
    const int h1 = (int)floorf(32.f * hi_h + 0.5f);
    const int w0 = (int)floorf(32.f * lo_w + 0.5f);
    const int w1 = (int)floorf(32.f * hi_w + 0.5f);
    *validOut = (h1 > h0) && (w1 > w0);
    const float sizeh = (float)(h1 - h0);
    const float steph = sizeh > 1.f ? (sizeh - 1.f) / (float)(SS - 1) : 0.f;
    const float ys = fminf(fmaxf((float)h0 + (float)sy * steph, 0.f), 31.f);
    const float sizew = (float)(w1 - w0);
    const float stepw = sizew > 1.f ? (sizew - 1.f) / (float)(SS - 1) : 0.f;
    const float xs = fminf(fmaxf((float)w0 + (float)sx * stepw, 0.f), 31.f);
    const int y0 = (int)floorf(ys);
    const int y1 = y0 + 1 > 31 ? 31 : y0 + 1;
    const float wy = ys - (float)y0;
    const int x0 = (int)floorf(xs);
    const int x1 = x0 + 1 > 31 ? 31 : x0 + 1;
    const float wx = xs - (float)x0;
    const int xa = flip ? 31 - x0 : x0;     // flip along W happens before crop
    const int xb = flip ? 31 - x1 : x1;
    const float v00 = base[y0 * 32 + xa], v01 = base[y0 * 32 + xb];
    const float v10 = base[y1 * 32 + xa], v11 = base[y1 * 32 + xb];
    return (1.f - wy) * ((1.f - wx) * v00 + wx * v01) +
           wy * ((1.f - wx) * v10 + wx * v11);
}

__global__ __launch_bounds__(128) void var_prep(const float* __restrict__ qv,
                                                const float* __restrict__ kv,
                                                const float* __restrict__ mq,
                                                const float* __restrict__ mk,
                                                _Float16* __restrict__ Apack,
                                                float* __restrict__ outVar) {
    const int row = blockIdx.x;                 // b*256 + sy*16 + sx
    const int b  = row >> 8;
    const int sy = (row >> 4) & 15;
    const int sx = row & 15;
    const int c  = threadIdx.x;

    const float iq = mq[b*5+0], jq = mq[b*5+1], hq = mq[b*5+2], wq = mq[b*5+3], fq = mq[b*5+4];
    const float ik = mk[b*5+0], jk = mk[b*5+1], hk = mk[b*5+2], wk = mk[b*5+3], fk = mk[b*5+4];
    const float bh0 = fmaxf(iq, ik),       bw0 = fmaxf(jq, jk);
    const float bh1 = fminf(iq+hq, ik+hk), bw1 = fminf(jq+wq, jk+wk);

    int vq, vk;
    const float* qb = qv + (size_t)(b * DIMC + c) * 1024;
    const float* kb = kv + (size_t)(b * DIMC + c) * 1024;
    const float qs = sample_map(qb, fq > 0.5f, (bh0-iq)/hq, (bh1-iq)/hq,
                                (bw0-jq)/wq, (bw1-jq)/wq, sy, sx, &vq);
    const float ks = sample_map(kb, fk > 0.5f, (bh0-ik)/hk, (bh1-ik)/hk,
                                (bw0-jk)/wk, (bw1-jk)/wk, sy, sx, &vk);
    const float valid = (vq && vk) ? 1.f : 0.f;

    __shared__ float s1[128], s2[128];
    s1[c] = qs * qs; s2[c] = ks * ks;
    __syncthreads();
    for (int off = 64; off > 0; off >>= 1) {
        if (c < off) { s1[c] += s1[c + off]; s2[c] += s2[c + off]; }
        __syncthreads();
    }
    const float qn = qs / fmaxf(sqrtf(s1[0]), 1e-12f) * valid;
    const float kn = ks / fmaxf(sqrtf(s2[0]), 1e-12f) * valid;
    __syncthreads();
    s1[c] = qn * kn;
    __syncthreads();
    for (int off = 64; off > 0; off >>= 1) {
        if (c < off) s1[c] += s1[c + off];
        __syncthreads();
    }
    if (c == 0) outVar[(size_t)row * OUTW] = s1[0] * INV_T;

    // Scatter qn into A-fragment layout (16x32 f16 tile per (mt,kb)).
    // Contiguous K order (matches B pack): lanes 0-15 hold K 0..15, lanes 16-31 hold
    // K 16..31; m = lane%16.
    const int kk = c & 31, kbI = c >> 5;
    const int g = kk >> 4;
    const int i = kk & 15;
    const int mt = row >> 4;
    const int lane = (row & 15) + (g << 4);
    Apack[(((mt * 4 + kbI) * 32 + lane) << 4) + i] = (_Float16)qn;
}

// ---------------- big GEMM: q_all (2048x128) @ queue_var (128x65536) ---------
// 8 waves / block, each wave owns 16(M) x 64(N); block tile 32 x 256.
__global__ __launch_bounds__(256) void var_gemm(const _Float16* __restrict__ Apack,
                                                const _Float16* __restrict__ Bpack,
                                                float* __restrict__ outVar) {
    const int lane = threadIdx.x & 31;
    const int wave = threadIdx.x >> 5;
    const int wm = wave & 1;
    const int wn = wave >> 1;
    const int mt  = blockIdx.y * 2 + wm;        // 16-row fragment index (0..127)
    const int nb0 = blockIdx.x * 16 + wn * 4;   // 16-col fragment index

    v8f acc0 = {}, acc1 = {}, acc2 = {}, acc3 = {};
#pragma unroll
    for (int kb = 0; kb < 4; ++kb) {
        const v16h a  = *(const v16h*)(Apack + (((mt * 4 + kb) * 32 + lane) << 4));
        const v16h b0 = *(const v16h*)(Bpack + ((((nb0 + 0) * 4 + kb) * 32 + lane) << 4));
        const v16h b1 = *(const v16h*)(Bpack + ((((nb0 + 1) * 4 + kb) * 32 + lane) << 4));
        const v16h b2 = *(const v16h*)(Bpack + ((((nb0 + 2) * 4 + kb) * 32 + lane) << 4));
        const v16h b3 = *(const v16h*)(Bpack + ((((nb0 + 3) * 4 + kb) * 32 + lane) << 4));
        acc0 = __builtin_amdgcn_wmma_f32_16x16x32_f16(false, a, false, b0, (short)0, acc0, false, false);
        acc1 = __builtin_amdgcn_wmma_f32_16x16x32_f16(false, a, false, b1, (short)0, acc1, false, false);
        acc2 = __builtin_amdgcn_wmma_f32_16x16x32_f16(false, a, false, b2, (short)0, acc2, false, false);
        acc3 = __builtin_amdgcn_wmma_f32_16x16x32_f16(false, a, false, b3, (short)0, acc3, false, false);
    }

    // C/D layout: n = lane%16; VGPR r holds row m = r + 8*(lane/16).
    const size_t m0 = (size_t)(mt * 16 + (lane >> 4) * 8);
    const int nlo = lane & 15;
#define STORE_TILE(ACC, T)                                                     \
    {                                                                          \
        float* p = outVar + m0 * OUTW + 1 + (size_t)((nb0 + (T)) * 16 + nlo);  \
        _Pragma("unroll") for (int r = 0; r < 8; ++r)                          \
            p[(size_t)r * OUTW] = ACC[r] * INV_T;                              \
    }
    STORE_TILE(acc0, 0)
    STORE_TILE(acc1, 1)
    STORE_TILE(acc2, 2)
    STORE_TILE(acc3, 3)
#undef STORE_TILE
}

// ---------------------------------------------------------------------------

extern "C" void kernel_launch(void* const* d_in, const int* in_sizes, int n_in,
                              void* d_out, int out_size, void* d_ws, size_t ws_size,
                              hipStream_t stream) {
    (void)in_sizes; (void)n_in; (void)out_size; (void)ws_size;
    const float* q_inv     = (const float*)d_in[0];
    const float* k_inv     = (const float*)d_in[1];
    const float* q_var     = (const float*)d_in[2];
    const float* k_var     = (const float*)d_in[3];
    const float* meta_q    = (const float*)d_in[4];
    const float* meta_k    = (const float*)d_in[5];
    const float* queue_inv = (const float*)d_in[6];
    const float* queue_var = (const float*)d_in[7];
    float* out = (float*)d_out;

    // Workspace layout (~17.3 MB total):
    char* ws = (char*)d_ws;
    _Float16* Apack = (_Float16*)ws;                 // 2048*128 f16  = 512 KB
    float*    qinvn = (float*)(ws + 524288);         // 8*128 f32     = 4 KB
    _Float16* Bpack = (_Float16*)(ws + 528384);      // 128*65536 f16 = 16 MB

    float* outVar = out + (size_t)8 * OUTW;          // logits_var block

    inv_prep<<<8, 128, 0, stream>>>(q_inv, k_inv, qinvn, out);
    inv_neg<<<KQ / 256, 256, 0, stream>>>(qinvn, queue_inv, out);
    pack_queue<<<(DIMC * KQ) / 256, 256, 0, stream>>>(queue_var, Bpack);
    var_prep<<<2048, 128, 0, stream>>>(q_var, k_var, meta_q, meta_k, Apack, outVar);
    var_gemm<<<dim3(KQ / 256, 2048 / 32), 256, 0, stream>>>(Apack, Bpack, outVar);
}
